// Model_33809982554284
// MI455X (gfx1250) — matrix operand order repaired and checked by reading, not compiled, not run on hardware
//
#include <hip/hip_runtime.h>

// ---------------------------------------------------------------------------
// CDNA5 (gfx1250) fused GNN autoencoder:
//  - one persistent workgroup per batch, activations live in LDS (264 KB slab)
//  - bf16 V_WMMA_F32_16X16X32_BF16 for both dense and graph-propagation GEMMs
//  - 2x2 register tile blocking: every A/B fragment feeds two WMMAs
//  - weights pre-transposed to bf16 [O][K] so the B-operand load is contiguous
//  - dense output stored transposed in LDS so it feeds prop's B operand directly
// ---------------------------------------------------------------------------

typedef __attribute__((ext_vector_type(16))) __bf16 v16bf;
typedef __attribute__((ext_vector_type(8)))  __bf16 v8bf;
typedef __attribute__((ext_vector_type(8)))  float  v8f;

#define NODES   255     // real node count (17*15)
#define NPAD    256     // padded node count (multiple of 16)
#define NSTR    264     // padded row stride (NPAD+8) -> breaks LDS bank conflicts
#define BATCH   256

// layer tables: K (input dim), Dout (real output dim), OP (padded output dim)
__device__ __constant__ int d_K[12]    = {64,64,64,128,128,256,256,256,256,128,128,64};
__device__ __constant__ int d_Dout[12] = {64,64,128,128,256,256,256,256,128,128,64,2};
__device__ __constant__ int d_OP[12]   = {64,64,128,128,256,256,256,256,128,128,64,16};
__device__ __constant__ int d_wtoff[12]= {0,4096,8192,16384,32768,65536,131072,196608,
                                          262144,294912,311296,319488};
// dad selection: enc even->s_dadsm(0), enc odd->t_dadsm(1), dec even->t_dadsp(2), dec odd->s_dadsp(3)
__device__ __constant__ int d_dadIdx[12]   = {0,1,0,1,0,1,2,3,2,3,2,3};
// scale selection: 0 -> scale_s, 1 -> scale_t
__device__ __constant__ int d_scaleSel[12] = {0,1,0,1,0,1,1,0,1,0,1,0};

struct MainParams {
  const float*  H;         // [256,255,64] fp32
  float*        out;       // [256,255,2]  fp32
  const __bf16* wt;        // packed transposed bf16 weights (see d_wtoff)
  const __bf16* dad[4];    // bf16 padded [256][264] adjacencies
  const float*  scale_s;   // [256][256]
  const float*  scale_t;   // [256][256]
  const float*  bias[12];  // fp32 bias arrays (real length Dout[l])
};

// --- setup: W [K][Dout] fp32 -> Wt [OP][K] bf16 (transposed, zero-padded) ---
__global__ void k_convert_w(const float* __restrict__ W, __bf16* __restrict__ Wt,
                            int K, int Dout, int OP) {
  int e = blockIdx.x * blockDim.x + threadIdx.x;
  if (e >= OP * K) return;
  int o = e / K, k = e - o * K;
  float v = (o < Dout) ? W[(size_t)k * Dout + o] : 0.f;
  Wt[e] = (__bf16)v;
}

// --- setup: dad [255][255] fp32 -> [256][264] bf16, zero padded row/col ---
__global__ void k_convert_dad(const float* __restrict__ src, __bf16* __restrict__ dst) {
  int e = blockIdx.x * blockDim.x + threadIdx.x;
  if (e >= NPAD * NSTR) return;
  int r = e / NSTR, c = e - r * NSTR;
  float v = (r < NODES && c < NODES) ? src[(size_t)r * NODES + c] : 0.f;
  dst[e] = (__bf16)v;
}

// --- setup: per-(b,n) propagation scales.
// scale = (dy*(1-dz)+dz) * row_sum / (dy*row_sum) == (dy*(1-dz)+dz)/dy  (row_sum cancels)
__global__ void k_scale(const float* __restrict__ dz_in,
                        const float* sw, const float* sb,
                        const float* tw, const float* tb,
                        float* __restrict__ ss, float* __restrict__ st) {
  int e = blockIdx.x * blockDim.x + threadIdx.x; // b*256+n
  if (e >= BATCH * NPAD) return;
  int b = e >> 8, n = e & 255;
  float dz = (n < NODES) ? dz_in[(size_t)b * NODES + n] : 1.f;  // pad node: identity
  float cs = sw[0] + sb[0];
  float ct = tw[0] + tb[0];
  float dys = cs * (1.f - dz) + dz;
  float dyt = ct * (1.f - dz) + dz;
  ss[e] = (dys * (1.f - dz) + dz) / dys;
  st[e] = (dyt * (1.f - dz) + dz) / dyt;
}

// fragment loaders in this code's contiguous k order (not the ISA table order; A and B match):
// A 16x32: lane holds row M=L&15, 16 consecutive K at (L>>4)*16 -> two 16B loads
// B 32x16: lane holds col N=L&15, 16 consecutive K at (L>>4)*16 (operand stored [N][K])
__device__ __forceinline__ v16bf ldfragA(const __bf16* p) {
  v8bf lo = *(const v8bf*)(p);
  v8bf hi = *(const v8bf*)(p + 8);
  v16bf r;
  #pragma unroll
  for (int i = 0; i < 8; ++i) { r[i] = lo[i]; r[i+8] = hi[i]; }
  return r;
}
__device__ __forceinline__ v16bf ldfragB(const __bf16* p) {
  v8bf lo = *(const v8bf*)(p);
  v8bf hi = *(const v8bf*)(p + 8);
  v16bf r;
  #pragma unroll
  for (int i = 0; i < 8; ++i) { r[i] = lo[i]; r[i+8] = hi[i]; }
  return r;
}
__device__ __forceinline__ v8f wmma_bf16(v16bf a, v16bf b, v8f c) {
  return __builtin_amdgcn_wmma_f32_16x16x32_bf16(false, a, false, b,
                                                 (short)0, c, false, false);
}

// ---------------------------------------------------------------------------
// Fused main kernel: 256 blocks (one per batch) x 256 threads (8 wave32s).
// LDS: sX [node][k] (stride K+8), sY [o][node] (stride 264), ping-pong.
// ---------------------------------------------------------------------------
__global__ __launch_bounds__(256)
void k_gnn_fused(MainParams p) {
  __shared__ __bf16 sX[NPAD * NSTR];   // 135168 B
  __shared__ __bf16 sY[NPAD * NSTR];   // 135168 B  (total 264 KB of 320 KB/WGP)

  const int b    = blockIdx.x;
  const int tid  = threadIdx.x;
  const int lane = tid & 31;
  const int wave = tid >> 5;
  const int lrow = lane & 15;       // A: M row / B: N col within tile
  const int lgrp = lane >> 4;       // lane half -> K sub-block select

  // stage H[b] (fp32 [255][64]) into sX as bf16, stride 72, zero-padded
  for (int e = tid; e < NPAD * 72; e += 256) {
    int r = e / 72, c = e - r * 72;
    float v = (r < NODES && c < 64) ? p.H[((size_t)b * NODES + r) * 64 + c] : 0.f;
    sX[e] = (__bf16)v;
  }
  __syncthreads();

  for (int l = 0; l < 12; ++l) {
    const int K    = d_K[l];
    const int Dout = d_Dout[l];
    const int OP   = d_OP[l];
    const int sx   = K + 8;                 // sX row stride this layer
    const int tilesN  = OP >> 4;
    const int tilesN2 = (tilesN + 1) >> 1;  // N-pairs (layer 11 clamps)
    const __bf16* Wt   = p.wt + d_wtoff[l];
    const float*  bias = p.bias[l];

    // ---- dense: sY[o][node] = sX[node][:] @ W[:, o] + bias[o] ----
    for (int t = wave; t < 8 * tilesN2; t += 8) {
      const int tm2 = t / tilesN2, tn2 = t - tm2 * tilesN2;
      const int tm0 = tm2 * 2, tm1 = tm0 + 1;
      const int tn0 = tn2 * 2;
      const int tn1 = (tn0 + 1 < tilesN) ? tn0 + 1 : tn0;
      v8f a00 = {0.f,0.f,0.f,0.f,0.f,0.f,0.f,0.f};
      v8f a01 = a00, a10 = a00, a11 = a00;
      const __bf16* ar0 = sX + (size_t)(tm0 * 16 + lrow) * sx + lgrp * 16;
      const __bf16* ar1 = ar0 + (size_t)16 * sx;
      const __bf16* br0 = Wt + (size_t)(tn0 * 16 + lrow) * K + lgrp * 16;
      const __bf16* br1 = Wt + (size_t)(tn1 * 16 + lrow) * K + lgrp * 16;
      for (int k0 = 0; k0 < K; k0 += 32) {
        v16bf A0 = ldfragA(ar0 + k0);
        v16bf A1 = ldfragA(ar1 + k0);
        v16bf B0 = ldfragB(br0 + k0);
        v16bf B1 = ldfragB(br1 + k0);
        a00 = wmma_bf16(A0, B0, a00);
        a01 = wmma_bf16(A0, B1, a01);
        a10 = wmma_bf16(A1, B0, a10);
        a11 = wmma_bf16(A1, B1, a11);
      }
      // epilogue: +bias, cvt bf16, one packed 16B ds_store per tile
      auto st_dense = [&](const v8f& acc, int tn, int tm) {
        const int o  = tn * 16 + lrow;
        const float bv = (o < Dout) ? bias[o] : 0.f;
        const int mb = tm * 16 + lgrp * 8;
        v8bf pk;
        #pragma unroll
        for (int v = 0; v < 8; ++v) pk[v] = (__bf16)(acc[v] + bv);
        *(v8bf*)(sY + (size_t)o * NSTR + mb) = pk;
      };
      st_dense(a00, tn0, tm0); st_dense(a01, tn1, tm0);
      st_dense(a10, tn0, tm1); st_dense(a11, tn1, tm1);
    }
    __syncthreads();

    // ---- prop: sX[node][o] = relu(scale[b,node] * sum_m dad[node][m] * sY[o][m]) ----
    const __bf16* dad = p.dad[d_dadIdx[l]];
    const float*  sc  = (d_scaleSel[l] ? p.scale_t : p.scale_s) + (size_t)b * NPAD;
    const int sz = OP + 8;                  // next layer's sX stride
    const bool last = (l == 11);
    for (int t = wave; t < 8 * tilesN2; t += 8) {
      const int tm2 = t / tilesN2, tn2 = t - tm2 * tilesN2;
      const int tm0 = tm2 * 2, tm1 = tm0 + 1;
      const int tn0 = tn2 * 2;
      const int tn1 = (tn0 + 1 < tilesN) ? tn0 + 1 : tn0;
      v8f a00 = {0.f,0.f,0.f,0.f,0.f,0.f,0.f,0.f};
      v8f a01 = a00, a10 = a00, a11 = a00;
      const __bf16* ar0 = dad + (size_t)(tm0 * 16 + lrow) * NSTR + lgrp * 16;
      const __bf16* ar1 = ar0 + (size_t)16 * NSTR;
      const __bf16* br0 = sY + (size_t)(tn0 * 16 + lrow) * NSTR + lgrp * 16;
      const __bf16* br1 = sY + (size_t)(tn1 * 16 + lrow) * NSTR + lgrp * 16;
      for (int k0 = 0; k0 < NPAD; k0 += 32) {
        __builtin_prefetch(ar0 + k0 + 32, 0, 1);   // global_prefetch next dad block
        v16bf A0 = ldfragA(ar0 + k0);
        v16bf A1 = ldfragA(ar1 + k0);
        v16bf B0 = ldfragB(br0 + k0);
        v16bf B1 = ldfragB(br1 + k0);
        a00 = wmma_bf16(A0, B0, a00);
        a01 = wmma_bf16(A0, B1, a01);
        a10 = wmma_bf16(A1, B0, a10);
        a11 = wmma_bf16(A1, B1, a11);
      }
      auto st_prop = [&](const v8f& acc, int tn, int tm) {
        const int o  = tn * 16 + lrow;
        const int mb = tm * 16 + lgrp * 8;
        const v8f sv = *(const v8f*)(sc + mb);   // 8 contiguous scales
        #pragma unroll
        for (int v = 0; v < 8; ++v) {
          const int node = mb + v;
          float z = acc[v] * sv[v];
          z = z > 0.f ? z : 0.f;
          sX[(size_t)node * sz + o] = (__bf16)z;
          if (last && o < 2 && node < NODES)
            p.out[((size_t)b * NODES + node) * 2 + o] = z;
        }
      };
      st_prop(a00, tn0, tm0); st_prop(a01, tn1, tm0);
      st_prop(a10, tn0, tm1); st_prop(a11, tn1, tm1);
    }
    __syncthreads();
  }
}

// ---------------------------------------------------------------------------
extern "C" void kernel_launch(void* const* d_in, const int* in_sizes, int n_in,
                              void* d_out, int out_size, void* d_ws, size_t ws_size,
                              hipStream_t stream) {
  (void)in_sizes; (void)n_in; (void)out_size; (void)ws_size;
  static const int hK[12]    = {64,64,64,128,128,256,256,256,256,128,128,64};
  static const int hDout[12] = {64,64,128,128,256,256,256,256,128,128,64,2};
  static const int hOP[12]   = {64,64,128,128,256,256,256,256,128,128,64,16};
  static const int hoff[12]  = {0,4096,8192,16384,32768,65536,131072,196608,
                                262144,294912,311296,319488};

  // workspace layout
  char*   ws        = (char*)d_ws;
  __bf16* wt        = (__bf16*)ws;                         // 320512 bf16 = 641 KB
  size_t  wt_bytes  = 320512 * sizeof(__bf16);
  size_t  dad_elems = (size_t)NPAD * NSTR;                 // 67584 per matrix
  __bf16* dadp      = (__bf16*)(ws + wt_bytes);            // 4 * 132 KB
  float*  scale_s   = (float*)(ws + wt_bytes + 4 * dad_elems * sizeof(__bf16));
  float*  scale_t   = scale_s + BATCH * NPAD;

  // 1) transpose + bf16-convert the 12 weight matrices (L2-resident afterwards)
  for (int l = 0; l < 12; ++l) {
    const float* W = (const float*)d_in[l < 6 ? 3 + l : 15 + (l - 6)];
    int n = hOP[l] * hK[l];
    k_convert_w<<<(n + 255) / 256, 256, 0, stream>>>(W, wt + hoff[l],
                                                     hK[l], hDout[l], hOP[l]);
  }
  // 2) bf16-convert + zero-pad the 4 adjacency matrices
  static const int dadInputIdx[4] = {31, 32, 33, 34};  // s_dadsm, t_dadsm, s_dadsp, t_dadsp
  for (int i = 0; i < 4; ++i)
    k_convert_dad<<<((int)(NPAD * NSTR) + 255) / 256, 256, 0, stream>>>(
        (const float*)d_in[dadInputIdx[i]], dadp + (size_t)i * dad_elems);
  // 3) propagation scales (row_sum cancels analytically; computed faithfully per (b,n))
  k_scale<<<(BATCH * NPAD) / 256, 256, 0, stream>>>(
      (const float*)d_in[2],
      (const float*)d_in[27], (const float*)d_in[28],
      (const float*)d_in[29], (const float*)d_in[30],
      scale_s, scale_t);

  // 4) fused 12-layer network, one workgroup per batch, activations in LDS
  MainParams p;
  p.H   = (const float*)d_in[0];
  p.out = (float*)d_out;
  p.wt  = wt;
  for (int i = 0; i < 4; ++i) p.dad[i] = dadp + (size_t)i * dad_elems;
  p.scale_s = scale_s;
  p.scale_t = scale_t;
  for (int l = 0; l < 12; ++l)
    p.bias[l] = (const float*)d_in[l < 6 ? 9 + l : 21 + (l - 6)];

  k_gnn_fused<<<BATCH, 256, 0, stream>>>(p);
}
